// Contrast_MultiHeadSelfAttention_22119081574883
// MI455X (gfx1250) — matrix operand order repaired and checked by reading, not compiled, not run on hardware
//
#include <hip/hip_runtime.h>
#include <stdint.h>

#define DIM 1024

typedef __attribute__((ext_vector_type(16))) __bf16 v16bf;
typedef __attribute__((ext_vector_type(8)))  float  v8f;

union Frag {
  uint4 q[2];
  unsigned int u[8];
  v16bf v;
};

// Packed f32x2 -> bf16x2 in one register via CDNA5 V_FMA_MIXLO/MIXHI_BF16
// (VOP3P opcodes 62/63; RNE conversion of fma(a,1.0,0)).
__device__ __forceinline__ unsigned int pk_bf16(float a, float b) {
  unsigned int r;
  asm("v_fma_mixlo_bf16 %0, %1, 1.0, 0\n\t"
      "v_fma_mixhi_bf16 %0, %2, 1.0, 0"
      : "=&v"(r)
      : "v"(a), "v"(b));
  return r;
}

__device__ __forceinline__ unsigned short f2bf(float a) {
  unsigned int r;
  asm("v_fma_mixlo_bf16 %0, %1, 1.0, 0" : "=v"(r) : "v"(a));
  return (unsigned short)r;
}

// Fast reciprocal: single v_rcp_f32 (TRANS unit, co-executes with XDL WMMA)
__device__ __forceinline__ float frcp(float x) { return __builtin_amdgcn_rcpf(x); }

// ---------------------------------------------------------------------------
// Projection GEMM:  out = X @ W^T  (X: [4096,1024] f32, W: [1024,1024] f32
// row-major (out_feat, in_feat)).  Output bf16:
//   transposed==0 : out[row*1024 + col]                       (Q, K)
//   transposed==1 : out[(batch*1024 + col)*1024 + (row&1023)] (V^T per batch)
// Block tile 128x128, K-step 32, 8 waves (4x2), each wave 32x64 (8 WMMA/step).
// ---------------------------------------------------------------------------
__global__ __launch_bounds__(256) void proj_bf16_kernel(
    const float* __restrict__ X, const float* __restrict__ W,
    unsigned short* __restrict__ out, int transposed) {
  __shared__ __align__(16) unsigned int As[128 * 16];  // 128 rows x 32 bf16
  __shared__ __align__(16) unsigned int Bs[128 * 16];

  const int tid  = threadIdx.x;
  const int lane = tid & 31;
  const int wid  = tid >> 5;
  const int m0 = blockIdx.x * 128;
  const int n0 = blockIdx.y * 128;
  const int wm = wid >> 1;       // 0..3 (M)
  const int wn = wid & 1;        // 0..1 (N)
  const int lm = lane & 15;
  const int lh = lane >> 4;

  v8f zero = {};
  v8f acc[2][4];
#pragma unroll
  for (int i = 0; i < 2; ++i)
#pragma unroll
    for (int j = 0; j < 4; ++j) acc[i][j] = zero;

  const int r    = tid >> 1;     // 0..127 (LDS fill row)
  const int half = tid & 1;      // which 16 of the 32 K columns

  for (int kt = 0; kt < 32; ++kt) {
    const int k0 = kt * 32;
    {
      const float* xa = X + (size_t)(m0 + r) * DIM + k0 + half * 16;
      const float* xb = W + (size_t)(n0 + r) * DIM + k0 + half * 16;
      if (kt + 1 < 32) {                       // prefetch next K tile
        __builtin_prefetch(xa + 32, 0, 0);
        __builtin_prefetch(xb + 32, 0, 0);
      }
      const float4* sa = (const float4*)xa;
      const float4* sb = (const float4*)xb;
#pragma unroll
      for (int i = 0; i < 4; ++i) {
        float4 a = sa[i];
        float4 b = sb[i];
        As[r * 16 + half * 8 + i * 2 + 0] = pk_bf16(a.x, a.y);
        As[r * 16 + half * 8 + i * 2 + 1] = pk_bf16(a.z, a.w);
        Bs[r * 16 + half * 8 + i * 2 + 0] = pk_bf16(b.x, b.y);
        Bs[r * 16 + half * 8 + i * 2 + 1] = pk_bf16(b.z, b.w);
      }
    }
    __syncthreads();

    Frag a[2], b[4];
#pragma unroll
    for (int mi = 0; mi < 2; ++mi) {     // A frag: lane m, K = lh*16 + 2v
      int m = wm * 32 + mi * 16 + lm;
      a[mi].q[0] = *(const uint4*)&As[m * 16 + lh * 8];
      a[mi].q[1] = *(const uint4*)&As[m * 16 + lh * 8 + 4];
    }
#pragma unroll
    for (int ni = 0; ni < 4; ++ni) {     // B frag: lane n, K = lh*16 + 2v
      int n = wn * 64 + ni * 16 + lm;
      b[ni].q[0] = *(const uint4*)&Bs[n * 16 + lh * 8];
      b[ni].q[1] = *(const uint4*)&Bs[n * 16 + lh * 8 + 4];
    }
#pragma unroll
    for (int mi = 0; mi < 2; ++mi)
#pragma unroll
      for (int ni = 0; ni < 4; ++ni)
        acc[mi][ni] = __builtin_amdgcn_wmma_f32_16x16x32_bf16(
            false, a[mi].v, false, b[ni].v, (short)0, acc[mi][ni], false, false);
    __syncthreads();
  }

#pragma unroll
  for (int mi = 0; mi < 2; ++mi)
#pragma unroll
    for (int ni = 0; ni < 4; ++ni) {
      int col = n0 + wn * 64 + ni * 16 + lm;
#pragma unroll
      for (int rr = 0; rr < 8; ++rr) {
        int row = m0 + wm * 32 + mi * 16 + rr + 8 * lh;
        unsigned short hv = f2bf(acc[mi][ni][rr]);
        size_t addr;
        if (transposed) {
          int bb = row >> 10, t = row & 1023;
          addr = ((size_t)(bb * DIM + col)) * DIM + t;
        } else {
          addr = (size_t)row * DIM + col;
        }
        out[addr] = hv;
      }
    }
}

// ---------------------------------------------------------------------------
// Fused contrastive attention. One wave handles 16 query rows of one (b,h).
// Pass 1: online (rowmax, Z) over all 1024 keys via score WMMAs.
// Pass 2: recompute score tiles, d = exp(s-m)/Z, w2 = exp(d), wa = rcp(w2),
//         transpose weights through LDS to A-fragments, accumulate both
//         P@V branches (vT gives contiguous K=token B-fragments).
// ---------------------------------------------------------------------------
__global__ __launch_bounds__(128) void attn_kernel(
    const unsigned short* __restrict__ qb, const unsigned short* __restrict__ kb,
    const unsigned short* __restrict__ vT,
    float* __restrict__ c_att, float* __restrict__ att) {
  __shared__ __align__(16) unsigned short wls[4][2][16 * 32];  // per-wave w2/wa tiles

  const int lane = threadIdx.x & 31;
  const int wid  = threadIdx.x >> 5;
  const int task = blockIdx.x * 4 + wid;     // 4096 tasks
  const int bh   = task >> 6;
  const int qblk = task & 63;
  const int b = bh >> 4;
  const int h = bh & 15;
  const int lm = lane & 15;
  const int lh = lane >> 4;

  const unsigned int* qu = (const unsigned int*)qb;
  const unsigned int* ku = (const unsigned int*)kb;
  const unsigned int* vu = (const unsigned int*)vT;

  const int token0 = b * 1024 + qblk * 16;   // flat query row base
  const int f0 = h * 64;                     // head feature base

  // Q A-fragments for the two dk K-steps (resident all kernel)
  Frag aq[2];
#pragma unroll
  for (int s = 0; s < 2; ++s) {
    int tok = token0 + lm;
    const unsigned int* p = qu + (size_t)tok * 512 + ((f0 + s * 32) >> 1) + lh * 8;
    aq[s].q[0] = *(const uint4*)p;
    aq[s].q[1] = *(const uint4*)(p + 4);
  }

  // ---------------- pass 1: row max / Z ----------------
  float mrow[8], zrow[8];
#pragma unroll
  for (int r = 0; r < 8; ++r) { mrow[r] = -1e30f; zrow[r] = 0.0f; }

  for (int jt = 0; jt < 64; ++jt) {
    Frag bk[2];
#pragma unroll
    for (int s = 0; s < 2; ++s) {
      int tokk = b * 1024 + jt * 16 + lm;
      const unsigned int* p = ku + (size_t)tokk * 512 + ((f0 + s * 32) >> 1) + lh * 8;
      bk[s].q[0] = *(const uint4*)p;
      bk[s].q[1] = *(const uint4*)(p + 4);
    }
    if (jt + 1 < 64) {                       // prefetch next key tile
      int tokn = b * 1024 + (jt + 1) * 16 + lm;
      __builtin_prefetch((const void*)(ku + (size_t)tokn * 512 + (f0 >> 1) + lh * 8), 0, 0);
    }
    v8f c = {};
    c = __builtin_amdgcn_wmma_f32_16x16x32_bf16(false, aq[0].v, false, bk[0].v, (short)0, c, false, false);
    c = __builtin_amdgcn_wmma_f32_16x16x32_bf16(false, aq[1].v, false, bk[1].v, (short)0, c, false, false);
#pragma unroll
    for (int r = 0; r < 8; ++r) {
      float sv = c[r] * 0.125f;            // 1/sqrt(64)
      float mn = fmaxf(mrow[r], sv);
      zrow[r] = zrow[r] * __expf(mrow[r] - mn) + __expf(sv - mn);
      mrow[r] = mn;
    }
  }
  // merge (m,Z) across the 16 lanes that share each row
#pragma unroll
  for (int r = 0; r < 8; ++r) {
#pragma unroll
    for (int mask = 1; mask < 16; mask <<= 1) {
      float om = __shfl_xor(mrow[r], mask, 32);
      float oz = __shfl_xor(zrow[r], mask, 32);
      float mn = fmaxf(mrow[r], om);
      zrow[r] = zrow[r] * __expf(mrow[r] - mn) + oz * __expf(om - mn);
      mrow[r] = mn;
    }
  }
  float invz[8];
#pragma unroll
  for (int r = 0; r < 8; ++r) invz[r] = frcp(zrow[r]);

  // ---------------- pass 2: weights + both P@V branches ----------------
  v8f zero = {};
  v8f u2[4], ua[4];
#pragma unroll
  for (int ft = 0; ft < 4; ++ft) { u2[ft] = zero; ua[ft] = zero; }
  float z2[8], za[8];
#pragma unroll
  for (int r = 0; r < 8; ++r) { z2[r] = 0.0f; za[r] = 0.0f; }

  unsigned short* w2s = wls[wid][0];
  unsigned short* was = wls[wid][1];

  for (int jt2 = 0; jt2 < 32; ++jt2) {     // 32 key-tiles of width 32
#pragma unroll
    for (int sub = 0; sub < 2; ++sub) {
      int jt = jt2 * 2 + sub;
      Frag bk[2];
#pragma unroll
      for (int s = 0; s < 2; ++s) {
        int tokk = b * 1024 + jt * 16 + lm;
        const unsigned int* p = ku + (size_t)tokk * 512 + ((f0 + s * 32) >> 1) + lh * 8;
        bk[s].q[0] = *(const uint4*)p;
        bk[s].q[1] = *(const uint4*)(p + 4);
      }
      v8f c = {};
      c = __builtin_amdgcn_wmma_f32_16x16x32_bf16(false, aq[0].v, false, bk[0].v, (short)0, c, false, false);
      c = __builtin_amdgcn_wmma_f32_16x16x32_bf16(false, aq[1].v, false, bk[1].v, (short)0, c, false, false);
#pragma unroll
      for (int r = 0; r < 8; ++r) {
        float sv = c[r] * 0.125f;
        float d  = __expf(sv - mrow[r]) * invz[r];   // dist in [0,1]
        float w2 = __expf(d);                        // softmax(dist) numerator
        float wa = frcp(w2);                         // exp(-d): softmax(1-dist) numerator
        z2[r] += w2;
        za[r] += wa;
        int m = r + 8 * lh;
        int jloc = sub * 16 + lm;
        w2s[m * 32 + jloc] = f2bf(w2);
        was[m * 32 + jloc] = f2bf(wa);
      }
    }
    // re-load weights as A-fragments (cross-lane transpose via LDS)
    Frag aw2, awa;
    {
      const unsigned int* p2 = (const unsigned int*)w2s;
      const unsigned int* pa = (const unsigned int*)was;
      aw2.q[0] = *(const uint4*)&p2[lm * 16 + lh * 8];
      aw2.q[1] = *(const uint4*)&p2[lm * 16 + lh * 8 + 4];
      awa.q[0] = *(const uint4*)&pa[lm * 16 + lh * 8];
      awa.q[1] = *(const uint4*)&pa[lm * 16 + lh * 8 + 4];
    }
#pragma unroll
    for (int ft = 0; ft < 4; ++ft) {       // dv = 64 -> 4 feature tiles
      Frag bv;
      int feat = f0 + ft * 16 + lm;
      const unsigned int* p = vu + (size_t)(b * DIM + feat) * 512 + jt2 * 16 + lh * 8;
      if (ft == 0 && jt2 + 1 < 32)         // prefetch next V tile
        __builtin_prefetch((const void*)(p + 16), 0, 0);
      bv.q[0] = *(const uint4*)p;
      bv.q[1] = *(const uint4*)(p + 4);
      u2[ft] = __builtin_amdgcn_wmma_f32_16x16x32_bf16(false, aw2.v, false, bv.v, (short)0, u2[ft], false, false);
      ua[ft] = __builtin_amdgcn_wmma_f32_16x16x32_bf16(false, awa.v, false, bv.v, (short)0, ua[ft], false, false);
    }
  }

  // row sums of the two softmax denominators
#pragma unroll
  for (int r = 0; r < 8; ++r) {
#pragma unroll
    for (int mask = 1; mask < 16; mask <<= 1) {
      z2[r] += __shfl_xor(z2[r], mask, 32);
      za[r] += __shfl_xor(za[r], mask, 32);
    }
  }
  float rz2[8], rza[8];
#pragma unroll
  for (int r = 0; r < 8; ++r) { rz2[r] = frcp(z2[r]); rza[r] = frcp(za[r]); }

#pragma unroll
  for (int ft = 0; ft < 4; ++ft) {
    int col = f0 + ft * 16 + lm;
#pragma unroll
    for (int r = 0; r < 8; ++r) {
      int tok = token0 + r + 8 * lh;
      size_t addr = (size_t)tok * DIM + col;
      c_att[addr] = ua[ft][r] * rza[r];
      att[addr]   = u2[ft][r] * rz2[r];
    }
  }
}

// ---------------------------------------------------------------------------
extern "C" void kernel_launch(void* const* d_in, const int* in_sizes, int n_in,
                              void* d_out, int out_size, void* d_ws, size_t ws_size,
                              hipStream_t stream) {
  const float* x  = (const float*)d_in[0];
  const float* y  = (const float*)d_in[1];
  const float* Wq = (const float*)d_in[2];
  const float* Wk = (const float*)d_in[3];
  const float* Wv = (const float*)d_in[4];

  const size_t QKV = (size_t)4 * 1024 * 1024;   // elements per projected tensor
  unsigned short* qb = (unsigned short*)d_ws;   // 8 MB each (bf16)
  unsigned short* kb = qb + QKV;
  unsigned short* vT = kb + QKV;

  float* c_att = (float*)d_out;
  float* att   = c_att + QKV;

  dim3 pgrid(32, 8);   // 4096/128 x 1024/128
  proj_bf16_kernel<<<pgrid, 256, 0, stream>>>(x, Wq, qb, 0);
  proj_bf16_kernel<<<pgrid, 256, 0, stream>>>(x, Wk, kb, 0);
  proj_bf16_kernel<<<pgrid, 256, 0, stream>>>(y, Wv, vT, 1);

  attn_kernel<<<1024, 128, 0, stream>>>(qb, kb, vT, c_att, att);
}
